// MeshGraphNetDiffLoss_64278480552465
// MI455X (gfx1250) — matrix operand order repaired and checked by reading, not compiled, not run on hardware
//
#include <hip/hip_runtime.h>
#include <stdint.h>

typedef __attribute__((ext_vector_type(16))) _Float16 v16h;
typedef __attribute__((ext_vector_type(8)))  float    v8f;

#define NN_   20000
#define NE_   80000
#define H_    128
#define S_    3
#define NT_   9
#define T_    3
#define L_    15
#define NODE_NORMAL  0
#define NODE_DISABLE 2
#define NODE_INPUT   4
#define NODE_OUTPUT  5
#define NODE_WALL    6

static inline int cdiv_i(int a, int b) { return (a + b - 1) / b; }

// ---------------------------------------------------------------------------
// Weight conversion: W[K][Nout] fp32 -> Wt[NoutPad][Kpad] f16 (transposed, padded)
// ---------------------------------------------------------------------------
__global__ void k_transpose_w(const float* __restrict__ W, _Float16* __restrict__ Wt,
                              int K, int Nout, int Kpad, int NoutPad) {
  int i = blockIdx.x * blockDim.x + threadIdx.x;
  int total = Kpad * NoutPad;
  if (i >= total) return;
  int n = i / Kpad, k = i % Kpad;
  float v = (n < Nout && k < K) ? W[(size_t)k * Nout + n] : 0.0f;
  Wt[(size_t)n * Kpad + k] = (_Float16)v;
}

// ---------------------------------------------------------------------------
// WMMA GEMM: Y[M][ldY] = act(X[M][K] * Wt^T + bias).
// One wave = 32 rows x NTB*16 cols (two 16-row A tiles share each B tile).
// Per K-step: clause-load 2 A frags + all NTB B frags, single wait, then
// 2*NTB back-to-back v_wmma_f32_16x16x32_f16.
// X row-major f16 (K mult of 32), Wt = [Nout][K] f16 (pre-transposed).
// ---------------------------------------------------------------------------
template <int NTB>
__global__ __launch_bounds__(128) void
k_gemm(const _Float16* __restrict__ X, const _Float16* __restrict__ Wt,
       const float* __restrict__ bias, _Float16* __restrict__ Yh,
       float* __restrict__ Yf, int M, int K, int Ncap, int ldY, int relu) {
  const int wave = threadIdx.x >> 5;
  const int lane = threadIdx.x & 31;
  const int tile = blockIdx.x * 4 + wave;
  const int row0 = tile * 32;
  if (row0 >= M) return;  // wave-uniform; active waves keep EXEC all-ones
  const int lmod = lane & 15;
  const int lhi  = lane >> 4;

  const _Float16* arow0 = X + (size_t)(row0 + lmod) * K + (size_t)lhi * 8;
  const _Float16* arow1 = arow0 + (size_t)16 * K;

  v8f zero = {};
  v8f acc0[NTB], acc1[NTB];
#pragma unroll
  for (int nb = 0; nb < NTB; ++nb) { acc0[nb] = zero; acc1[nb] = zero; }

  union HF { v16h v; uint4 q[2]; };

  for (int k0 = 0; k0 < K; k0 += 32) {
    HF a0, a1;
    a0.q[0] = *(const uint4*)(arow0 + k0);        // K = k0+kb .. +7
    a0.q[1] = *(const uint4*)(arow0 + k0 + 16);   // K = k0+kb+16 .. +23
    a1.q[0] = *(const uint4*)(arow1 + k0);
    a1.q[1] = *(const uint4*)(arow1 + k0 + 16);
    HF b[NTB];
#pragma unroll
    for (int nb = 0; nb < NTB; ++nb) {
      const _Float16* brow = Wt + (size_t)(nb * 16 + lmod) * K + k0 + (size_t)lhi * 8;
      b[nb].q[0] = *(const uint4*)(brow);         // K = k0+kb .. +7
      b[nb].q[1] = *(const uint4*)(brow + 16);    // K = k0+kb+16 .. +23
    }
#pragma unroll
    for (int nb = 0; nb < NTB; ++nb) {
      acc0[nb] = __builtin_amdgcn_wmma_f32_16x16x32_f16(
          false, a0.v, false, b[nb].v, (short)0, acc0[nb], false, false);
      acc1[nb] = __builtin_amdgcn_wmma_f32_16x16x32_f16(
          false, a1.v, false, b[nb].v, (short)0, acc1[nb], false, false);
    }
  }

#pragma unroll
  for (int nb = 0; nb < NTB; ++nb) {
    int col = nb * 16 + lmod;
    float bv = (col < Ncap) ? bias[col] : 0.0f;
#pragma unroll
    for (int r = 0; r < 8; ++r) {
      // C/D layout: VGPR r -> M = r (+8 for hi lanes)
      int rowA = row0 + r + lhi * 8;
      int rowB = rowA + 16;
      float v0 = acc0[nb][r] + bv;
      float v1 = acc1[nb][r] + bv;
      if (relu) {
        v0 = v0 > 0.0f ? v0 : 0.0f;
        v1 = v1 > 0.0f ? v1 : 0.0f;
      }
      if (col < Ncap) {
        size_t o0 = (size_t)rowA * ldY + col;
        size_t o1 = (size_t)rowB * ldY + col;
        if (Yh) { Yh[o0] = (_Float16)v0; Yh[o1] = (_Float16)v1; }
        if (Yf) { Yf[o0] = v0; Yf[o1] = v1; }
      }
    }
  }
}

// ---------------------------------------------------------------------------
// LayerNorm over 128 features, one row per 32-lane group.
// ---------------------------------------------------------------------------
__global__ void k_layernorm(const float* __restrict__ Hin, const float* __restrict__ g,
                            const float* __restrict__ be, float* __restrict__ Yf,
                            _Float16* __restrict__ Yh, int M) {
  int row = (blockIdx.x * blockDim.x + threadIdx.x) >> 5;
  int lane = threadIdx.x & 31;
  if (row >= M) return;
  const float* h = Hin + (size_t)row * 128;
  int c = lane * 4;
  float4 x = *(const float4*)(h + c);
  float s = x.x + x.y + x.z + x.w;
  for (int m = 16; m >= 1; m >>= 1) s += __shfl_xor(s, m, 32);
  float mu = s * (1.0f / 128.0f);
  float dx = x.x - mu, dy = x.y - mu, dz = x.z - mu, dw = x.w - mu;
  float vs = dx * dx + dy * dy + dz * dz + dw * dw;
  for (int m = 16; m >= 1; m >>= 1) vs += __shfl_xor(vs, m, 32);
  float rinv = rsqrtf(vs * (1.0f / 128.0f) + 1e-5f);
  float4 gg = *(const float4*)(g + c);
  float4 bb = *(const float4*)(be + c);
  float o0 = dx * rinv * gg.x + bb.x;
  float o1 = dy * rinv * gg.y + bb.y;
  float o2 = dz * rinv * gg.z + bb.z;
  float o3 = dw * rinv * gg.w + bb.w;
  float* yo = Yf + (size_t)row * 128 + c;
  yo[0] = o0; yo[1] = o1; yo[2] = o2; yo[3] = o3;
  if (Yh) {
    _Float16* yh = Yh + (size_t)row * 128 + c;
    yh[0] = (_Float16)o0; yh[1] = (_Float16)o1;
    yh[2] = (_Float16)o2; yh[3] = (_Float16)o3;
  }
}

// ---------------------------------------------------------------------------
// Elementwise helpers
// ---------------------------------------------------------------------------
__global__ void k_zero(float* __restrict__ p, size_t n) {
  size_t i = (size_t)blockIdx.x * blockDim.x + threadIdx.x;
  if (i < n) p[i] = 0.0f;
}

__global__ void k_residual(float* __restrict__ Acc, const float* __restrict__ Delta,
                           _Float16* __restrict__ Ah, size_t total) {
  size_t i = (size_t)blockIdx.x * blockDim.x + threadIdx.x;
  if (i >= total) return;
  float v = Acc[i] + Delta[i];
  Acc[i] = v;
  Ah[i] = (_Float16)v;
}

// segment_sum: agg[r_idx[e]] += e_new[e]  (hardware f32 atomics)
__global__ void k_scatter(const float* __restrict__ Enew, const int* __restrict__ edges,
                          float* __restrict__ Agg, int ts) {
  int tid = blockIdx.x * blockDim.x + threadIdx.x;
  int e = tid >> 5;
  if (e >= NE_) return;
  int c = (tid & 31) * 4;
  int r = edges[((size_t)ts * NE_ + e) * 2 + 1];
  const float* src = Enew + (size_t)e * 128 + c;
  float* dst = Agg + (size_t)r * 128 + c;
  atomicAdd(dst + 0, src[0]);
  atomicAdd(dst + 1, src[1]);
  atomicAdd(dst + 2, src[2]);
  atomicAdd(dst + 3, src[3]);
}

// edgecat[e] = [E(e), V(s), V(r)]  (f16, 384 wide)
__global__ void k_concat_edge(const _Float16* __restrict__ Eh, const _Float16* __restrict__ Vh,
                              const int* __restrict__ edges, _Float16* __restrict__ Cat, int ts) {
  int tid = blockIdx.x * blockDim.x + threadIdx.x;
  int e = tid >> 5;
  if (e >= NE_) return;
  int c = (tid & 31) * 4;
  size_t eb = ((size_t)ts * NE_ + e) * 2;
  int s = edges[eb], r = edges[eb + 1];
  _Float16* out = Cat + (size_t)e * 384;
  *(uint2*)(out + c)       = *(const uint2*)(Eh + (size_t)e * 128 + c);
  *(uint2*)(out + 128 + c) = *(const uint2*)(Vh + (size_t)s * 128 + c);
  *(uint2*)(out + 256 + c) = *(const uint2*)(Vh + (size_t)r * 128 + c);
}

// nodecat[n] = [V(n), agg(n)]  (f16, 256 wide)
__global__ void k_concat_node(const _Float16* __restrict__ Vh, const float* __restrict__ Agg,
                              _Float16* __restrict__ Cat) {
  int tid = blockIdx.x * blockDim.x + threadIdx.x;
  int n = tid >> 5;
  if (n >= NN_) return;
  int c = (tid & 31) * 4;
  _Float16* out = Cat + (size_t)n * 256;
  *(uint2*)(out + c) = *(const uint2*)(Vh + (size_t)n * 128 + c);
  float4 a = *(const float4*)(Agg + (size_t)n * 128 + c);
  union { uint2 u; _Float16 h[4]; } pk;
  pk.h[0] = (_Float16)a.x; pk.h[1] = (_Float16)a.y;
  pk.h[2] = (_Float16)a.z; pk.h[3] = (_Float16)a.w;
  *(uint2*)(out + 128 + c) = pk.u;
}

// edge features: dist(2) + norm(1), normalized, padded to 32
__global__ void k_edge_feat(const float* __restrict__ mesh, const int* __restrict__ edges,
                            const float* __restrict__ em, const float* __restrict__ es,
                            _Float16* __restrict__ Xe, int ts) {
  int e = blockIdx.x * blockDim.x + threadIdx.x;
  if (e >= NE_) return;
  size_t eb = ((size_t)ts * NE_ + e) * 2;
  int s = edges[eb], r = edges[eb + 1];
  const float* mb = mesh + (size_t)ts * NN_ * 2;
  float dx = mb[(size_t)s * 2 + 0] - mb[(size_t)r * 2 + 0];
  float dy = mb[(size_t)s * 2 + 1] - mb[(size_t)r * 2 + 1];
  float nr = sqrtf(dx * dx + dy * dy);
  _Float16* row = Xe + (size_t)e * 32;
  row[0] = (_Float16)((dx - em[0]) / (es[0] + 1e-8f));
  row[1] = (_Float16)((dy - em[1]) / (es[1] + 1e-8f));
  row[2] = (_Float16)((nr - em[2]) / (es[2] + 1e-8f));
#pragma unroll
  for (int i = 3; i < 32; ++i) row[i] = (_Float16)0.0f;
}

// node features: [state_hat(3), node_type(9)], normalized, padded to 32
__global__ void k_node_feat(const float* __restrict__ scur, const float* __restrict__ ntype,
                            const float* __restrict__ nm, const float* __restrict__ ns,
                            _Float16* __restrict__ Xn, int ts) {
  int n = blockIdx.x * blockDim.x + threadIdx.x;
  if (n >= NN_) return;
  _Float16* row = Xn + (size_t)n * 32;
  float f[12];
  f[0] = scur[(size_t)n * 3 + 0];
  f[1] = scur[(size_t)n * 3 + 1];
  f[2] = scur[(size_t)n * 3 + 2];
  const float* nt = ntype + ((size_t)ts * NN_ + n) * NT_;
#pragma unroll
  for (int j = 0; j < NT_; ++j) f[3 + j] = nt[j];
#pragma unroll
  for (int i = 0; i < 12; ++i)
    row[i] = (_Float16)((f[i] - nm[i]) / (ns[i] + 1e-8f));
#pragma unroll
  for (int i = 12; i < 32; ++i) row[i] = (_Float16)0.0f;
}

__device__ inline float hash_normal(uint32_t x) {
  x ^= x >> 16; x *= 0x7feb352du; x ^= x >> 15; x *= 0x846ca68bu; x ^= x >> 16;
  uint32_t y = x * 0x9e3779b9u + 0x85ebca6bu;
  y ^= y >> 16; y *= 0x7feb352du; y ^= y >> 15;
  float u1 = ((x >> 8) + 0.5f) * (1.0f / 16777216.0f);
  float u2 = ((y >> 8) + 0.5f) * (1.0f / 16777216.0f);
  return sqrtf(-2.0f * logf(u1)) * cosf(6.2831853f * u2);
}

// state0 = state[:,0] (+ noise if node-0 gate), also writes state_hat slot 0
__global__ void k_init_state(const float* __restrict__ state, const float* __restrict__ ntype,
                             float* __restrict__ scur, float* __restrict__ dout) {
  int n = blockIdx.x * blockDim.x + threadIdx.x;
  if (n >= NN_) return;
  // reference gate: nmask[:,0] -> depends only on node 0 at t=0
  bool bm = (ntype[NODE_NORMAL] == 1.0f) || (ntype[NODE_OUTPUT] == 1.0f);
#pragma unroll
  for (int s = 0; s < 3; ++s) {
    float v = state[(size_t)n * 3 + s];
    if (bm) v += 0.02f * hash_normal((uint32_t)(n * 3 + s + 1));
    scur[(size_t)n * 3 + s] = v;
    dout[(size_t)n * 3 + s] = v;  // state_hat[t=0]
  }
}

// decoder output -> next state / output_hat / target
__global__ void k_update(const float* __restrict__ dec, const float* __restrict__ state,
                         const float* __restrict__ ntype, const float* __restrict__ om,
                         const float* __restrict__ os, float* __restrict__ scur,
                         float* __restrict__ dout, int t) {
  int n = blockIdx.x * blockDim.x + threadIdx.x;
  if (n >= NN_) return;
  const float* st = state + ((size_t)t * NN_ + n) * 3;
  const float* nt = ntype + ((size_t)t * NN_ + n) * NT_;
  bool m = (nt[NODE_INPUT] == 1.0f) || (nt[NODE_WALL] == 1.0f) || (nt[NODE_DISABLE] == 1.0f);
  const size_t SH = (size_t)T_ * NN_ * 3;             // end of state_hat block
  const size_t OH = SH + (size_t)(T_ - 1) * NN_ * 3;  // end of output_hat block
#pragma unroll
  for (int s = 0; s < 3; ++s) {
    float o = dec[(size_t)n * 3 + s];
    float prev = scur[(size_t)n * 3 + s];
    float nxt = prev + o * os[s] + om[s];
    float tg = (st[s] - prev - om[s]) / (os[s] + 1e-8f);
    if (m) nxt = st[s];
    dout[((size_t)t * NN_ + n) * 3 + s] = nxt;                 // state_hat[t]
    dout[SH + ((size_t)(t - 1) * NN_ + n) * 3 + s] = o;        // output_hat[t-1]
    dout[OH + ((size_t)(t - 1) * NN_ + n) * 3 + s] = tg;       // target[t-1]
    scur[(size_t)n * 3 + s] = nxt;
  }
}

// ---------------------------------------------------------------------------
// Host side
// ---------------------------------------------------------------------------
struct MLPW {
  const float *W0, *W1, *W2, *b0, *b1, *b2, *g, *be;
  _Float16 *W0t, *W1t, *W2t;
  int K0, K0pad, N2, N2pad;
};

extern "C" void kernel_launch(void* const* d_in, const int* in_sizes, int n_in,
                              void* d_out, int out_size, void* d_ws, size_t ws_size,
                              hipStream_t stream) {
  (void)in_sizes; (void)n_in; (void)out_size; (void)ws_size;
  auto P = [&](int i) -> const float* { return (const float*)d_in[i]; };

  const float* mesh  = P(0);
  const int*   edges = (const int*)d_in[1];
  const float* state = P(2);
  const float* ntype = P(3);
  // d_in[4] = parameters (0 elements)
  // params leaves (jax tree, sorted keys): dec, fe, fv, gnn[i]{fe,fn}, norms
  const int PB = 5;
  const int GNN0 = PB + 22;                 // dec(6) + fe(8) + fv(8)
  const int NRM0 = GNN0 + 16 * L_;          // after 15 layers * (8+8)
  const float* ne_mean = P(NRM0 + 0);
  const float* ne_std  = P(NRM0 + 1);
  const float* nn_mean = P(NRM0 + 2);
  const float* nn_std  = P(NRM0 + 3);
  const float* no_mean = P(NRM0 + 4);
  const float* no_std  = P(NRM0 + 5);

  // workspace bump allocator (256B aligned)
  uint8_t* wsb = (uint8_t*)d_ws;
  size_t off = 0;
  auto alloc = [&](size_t bytes) -> void* {
    off = (off + 255) & ~(size_t)255;
    void* p = wsb + off;
    off += bytes;
    return p;
  };

  auto mkMLP = [&](int base, bool ln, int K0, int N2) -> MLPW {
    MLPW m{};
    m.W0 = P(base + 0); m.W1 = P(base + 1); m.W2 = P(base + 2);
    m.b0 = P(base + 3); m.b1 = P(base + 4); m.b2 = P(base + 5);
    if (ln) { m.be = P(base + 6); m.g = P(base + 7); }
    m.K0 = K0; m.K0pad = (K0 + 31) & ~31;
    m.N2 = N2; m.N2pad = (N2 + 15) & ~15;
    m.W0t = (_Float16*)alloc((size_t)128 * m.K0pad * 2);
    m.W1t = (_Float16*)alloc((size_t)128 * 128 * 2);
    m.W2t = (_Float16*)alloc((size_t)m.N2pad * 128 * 2);
    return m;
  };

  MLPW dec = mkMLP(PB + 0, false, 128, 3);
  MLPW fe  = mkMLP(PB + 6, true, 3, 128);
  MLPW fv  = mkMLP(PB + 14, true, 12, 128);
  MLPW gfe[L_], gfn[L_];
  for (int l = 0; l < L_; ++l) {
    gfe[l] = mkMLP(GNN0 + 16 * l + 0, true, 384, 128);
    gfn[l] = mkMLP(GNN0 + 16 * l + 8, true, 256, 128);
  }

  // activation buffers
  _Float16* Xnode   = (_Float16*)alloc((size_t)NN_ * 32 * 2);
  _Float16* Xedge   = (_Float16*)alloc((size_t)NE_ * 32 * 2);
  _Float16* Vh      = (_Float16*)alloc((size_t)NN_ * 128 * 2);
  _Float16* Eh      = (_Float16*)alloc((size_t)NE_ * 128 * 2);
  _Float16* edgecat = (_Float16*)alloc((size_t)NE_ * 384 * 2);
  _Float16* nodecat = (_Float16*)alloc((size_t)NN_ * 256 * 2);
  _Float16* h0h     = (_Float16*)alloc((size_t)NE_ * 128 * 2);
  _Float16* h1h     = (_Float16*)alloc((size_t)NE_ * 128 * 2);
  float* Vf   = (float*)alloc((size_t)NN_ * 128 * 4);
  float* Ef   = (float*)alloc((size_t)NE_ * 128 * 4);
  float* hpre = (float*)alloc((size_t)NE_ * 128 * 4);
  float* dnew = (float*)alloc((size_t)NE_ * 128 * 4);
  float* agg  = (float*)alloc((size_t)NN_ * 128 * 4);
  float* dout3= (float*)alloc((size_t)NN_ * 3 * 4);
  float* scur = (float*)alloc((size_t)NN_ * 3 * 4);

  float* out = (float*)d_out;

  // ---- weight conversion (weights ~4.6MB f16: stays L2-resident) ----
  auto conv = [&](const MLPW& m) {
    int t0 = 128 * m.K0pad;
    k_transpose_w<<<cdiv_i(t0, 256), 256, 0, stream>>>(m.W0, m.W0t, m.K0, 128, m.K0pad, 128);
    k_transpose_w<<<cdiv_i(128 * 128, 256), 256, 0, stream>>>(m.W1, m.W1t, 128, 128, 128, 128);
    int t2 = m.N2pad * 128;
    k_transpose_w<<<cdiv_i(t2, 256), 256, 0, stream>>>(m.W2, m.W2t, 128, m.N2, 128, m.N2pad);
  };
  conv(dec); conv(fe); conv(fv);
  for (int l = 0; l < L_; ++l) { conv(gfe[l]); conv(gfn[l]); }

  // ---- GEMM / MLP helpers ----
  auto gemm = [&](const _Float16* X, const _Float16* W, const float* b,
                  _Float16* Yhv, float* Yfv, int M, int K, int Nout, int Ncap,
                  int ldY, int relu) {
    int blocks = cdiv_i(M / 32, 4);  // 32-row tiles, 4 waves/block
    if (Nout == 128)
      k_gemm<8><<<blocks, 128, 0, stream>>>(X, W, b, Yhv, Yfv, M, K, Ncap, ldY, relu);
    else
      k_gemm<1><<<blocks, 128, 0, stream>>>(X, W, b, Yhv, Yfv, M, K, Ncap, ldY, relu);
  };
  auto mlp_ln = [&](const MLPW& m, const _Float16* X, int M, int K,
                    float* Yfv, _Float16* Yhv) {
    gemm(X, m.W0t, m.b0, h0h, nullptr, M, K, 128, 128, 128, 1);
    gemm(h0h, m.W1t, m.b1, h1h, nullptr, M, 128, 128, 128, 128, 1);
    gemm(h1h, m.W2t, m.b2, nullptr, hpre, M, 128, 128, 128, 128, 0);
    k_layernorm<<<cdiv_i(M * 32, 256), 256, 0, stream>>>(hpre, m.g, m.be, Yfv, Yhv, M);
  };

  // ---- forward ----
  k_init_state<<<cdiv_i(NN_, 256), 256, 0, stream>>>(state, ntype, scur, out);

  for (int t = 1; t < T_; ++t) {
    int ts = t - 1;
    k_edge_feat<<<cdiv_i(NE_, 256), 256, 0, stream>>>(mesh, edges, ne_mean, ne_std, Xedge, ts);
    k_node_feat<<<cdiv_i(NN_, 256), 256, 0, stream>>>(scur, ntype, nn_mean, nn_std, Xnode, ts);

    mlp_ln(fv, Xnode, NN_, 32, Vf, Vh);   // encoder: nodes
    mlp_ln(fe, Xedge, NE_, 32, Ef, Eh);   // encoder: edges

    for (int l = 0; l < L_; ++l) {
      k_concat_edge<<<cdiv_i(NE_ * 32, 256), 256, 0, stream>>>(Eh, Vh, edges, edgecat, ts);
      mlp_ln(gfe[l], edgecat, NE_, 384, dnew, nullptr);          // e_new (f32)
      k_zero<<<cdiv_i(NN_ * 128, 256), 256, 0, stream>>>(agg, (size_t)NN_ * 128);
      k_scatter<<<cdiv_i(NE_ * 32, 256), 256, 0, stream>>>(dnew, edges, agg, ts);
      k_residual<<<cdiv_i(NE_ * 128, 256), 256, 0, stream>>>(Ef, dnew, Eh, (size_t)NE_ * 128);
      k_concat_node<<<cdiv_i(NN_ * 32, 256), 256, 0, stream>>>(Vh, agg, nodecat); // uses old Vh
      mlp_ln(gfn[l], nodecat, NN_, 256, dnew, nullptr);          // v_new (f32)
      k_residual<<<cdiv_i(NN_ * 128, 256), 256, 0, stream>>>(Vf, dnew, Vh, (size_t)NN_ * 128);
    }

    // decoder (no LN), Nout padded 3->16, store only 3 cols
    gemm(Vh, dec.W0t, dec.b0, h0h, nullptr, NN_, 128, 128, 128, 128, 1);
    gemm(h0h, dec.W1t, dec.b1, h1h, nullptr, NN_, 128, 128, 128, 128, 1);
    gemm(h1h, dec.W2t, dec.b2, nullptr, dout3, NN_, 128, 16, 3, 3, 0);

    k_update<<<cdiv_i(NN_, 256), 256, 0, stream>>>(dout3, state, ntype, no_mean, no_std,
                                                   scur, out, t);
  }
}
